// MultiHeadedLocalAttention_12962211299911
// MI455X (gfx1250) — compile-verified, matrix operand order checked by reading, not run on hardware
//
#include <hip/hip_runtime.h>
#include <cstdint>

// ---------------------------------------------------------------------------
// Swin shifted-window attention for MI455X (gfx1250, wave32, WMMA + TDM).
// bf16 inputs to V_WMMA_F32_16X16X32_BF16, f32 accumulate.
// Workspace (bf16 = unsigned short):
//   q  : [B][64 win][12 head][49 tok][32 dh]
//   k  : [B][64 win][12 head][49 tok][32 dh]
//   v  : [B][64 win][12 head][32 dh][49 tok]   (pre-transposed for B-operand)
//   ab : [B][56][56][384]  attention output, shifted frame
// Total ws: 4 * 38,535,168 * 2 B ~= 294 MB.
// ---------------------------------------------------------------------------

typedef __attribute__((ext_vector_type(16))) unsigned short v16u;
typedef __attribute__((ext_vector_type(16))) __bf16         v16bf;
typedef __attribute__((ext_vector_type(8)))  float          v8f;
typedef __attribute__((ext_vector_type(4)))  unsigned int   u32x4;
typedef __attribute__((ext_vector_type(8)))  int            i32x8;
typedef __attribute__((ext_vector_type(4)))  int            i32x4;

#define DIMC  384
#define NHEAD 12
#define DHEAD 32
#define WSZ   7
#define HH    56
#define WWC   56
#define NB    32
#define NWIN  64
#define NTOK  49
#define MTOT  (NB*HH*WWC)     // 100352
#define NQKV  (3*NHEAD*DHEAD) // 1152
#define HEADSZ (NTOK*DHEAD)   // 1568 elements per (b,win,head)

static __device__ __forceinline__ unsigned short f2bf(float f) {
  unsigned int u = __float_as_uint(f);
  u += 0x7FFFu + ((u >> 16) & 1u);          // round-to-nearest-even
  return (unsigned short)(u >> 16);
}

// K-element index inside a 16-bit WMMA A/B operand (ISA 7.12.2).
static __device__ __forceinline__ int kmap(int e, int hf) {
  return (e < 8) ? (8 * hf + e) : (8 + 8 * hf + e);
}

static __device__ __forceinline__ v16bf frag_from_lds(const unsigned short* p, int hf) {
  v16u r;
#pragma unroll
  for (int e = 0; e < 16; ++e) r[e] = p[kmap(e, hf)];
  union { v16u u; v16bf b; } c; c.u = r;
  return c.b;
}

// --------------------------- TDM descriptor helpers ------------------------
// D# group 0 (ISA 8.3): count=1 | lds_addr | 57b global addr | type=2.
static __device__ __forceinline__ u32x4 tdm_g0(const void* lds_ptr, const void* gptr) {
  unsigned long long ga = (unsigned long long)gptr;
  unsigned lds = (unsigned)(unsigned long long)lds_ptr;   // addr[31:0] = LDS bytes
  u32x4 g;
  g[0] = 1u;                                              // count=1 (valid user D#)
  g[1] = lds;
  g[2] = (unsigned)ga;
  g[3] = ((unsigned)(ga >> 32) & 0x01FFFFFFu) | 0x80000000u; // type=2 ("image")
  return g;
}
// D# group 1 (ISA 8.4): data_size=1 (2B), 2D tile, dim0 stride; no pad/multicast.
static __device__ __forceinline__ i32x8 tdm_g1_2d(unsigned td0, unsigned td1,
                                                  unsigned tl0, unsigned tl1,
                                                  unsigned stride0) {
  i32x8 g;
  g[0] = (int)(1u << 16);                         // data_size = 2 bytes
  g[1] = (int)((td0 & 0xFFFFu) << 16);            // tensor_dim0[15:0] at bit48
  g[2] = (int)((td0 >> 16) | ((td1 & 0xFFFFu) << 16));
  g[3] = (int)((td1 >> 16) | (tl0 << 16));        // tile_dim0 at bit112
  g[4] = (int)(tl1 & 0xFFFFu);                    // tile_dim1 (tile_dim2 = 0)
  g[5] = (int)stride0;                            // tensor_dim0_stride[31:0]
  g[6] = 0;                                       // stride0 hi | stride1 lo
  g[7] = 0;                                       // stride1 hi
  return g;
}

// ---------------------------------------------------------------------------
// Kernel 1: qkv = shift(x) @ w_qkv   (M=100352, K=384, N=1152), bf16 WMMA.
// f32->bf16 conversion forces VALU staging here (TDM can't convert).
// ---------------------------------------------------------------------------
__global__ __launch_bounds__(256)
void qkv_kernel(const float* __restrict__ x, const float* __restrict__ w_qkv,
                unsigned short* __restrict__ qb, unsigned short* __restrict__ kbf,
                unsigned short* __restrict__ vb)
{
  __shared__ __align__(16) unsigned short As[64 * 32];   // [m][k]
  __shared__ __align__(16) unsigned short Bst[64 * 32];  // [n][k] (transposed)
  __shared__ int rowQK[64];                              // per-row dest bases
  __shared__ int rowV[64];

  const int m0 = blockIdx.y * 64;
  const int n0 = blockIdx.x * 64;
  const int tid = threadIdx.x;
  const int wave = tid >> 5, lane = tid & 31;
  const int idx = lane & 15, hf = lane >> 4;
  const int tj = wave & 3, tip = wave >> 2;

  // one thread per row: window/shift decomposition done once per block
  if (tid < 64) {
    int m = m0 + tid;
    int b = m / (HH * WWC); int rem = m % (HH * WWC);
    int y = rem / WWC, xx = rem % WWC;
    int win = (y / WSZ) * 8 + (xx / WSZ);
    int tok = (y % WSZ) * WSZ + (xx % WSZ);
    int base = (b * NWIN + win) * (NHEAD * HEADSZ);
    rowQK[tid] = base + tok * DHEAD;
    rowV[tid]  = base + tok;
  }

  // hoist loop-invariant source pointers (shift folded in)
  const int mloc = tid >> 2, kg = tid & 3;
  const float* arow;
  {
    int mg = m0 + mloc;
    int b = mg / (HH * WWC); int rem = mg % (HH * WWC);
    int y = rem / WWC, xx = rem % WWC;
    arow = x + (((size_t)b * HH + (y + 3) % HH) * WWC + (xx + 3) % WWC) * DIMC + kg * 8;
  }
  const int krow = tid >> 3, ng = tid & 7;
  const float* brow = w_qkv + (size_t)krow * NQKV + n0 + ng * 8;

  v8f acc0 = {}; v8f acc1 = {};

  for (int kb = 0; kb < DIMC / 32; ++kb) {
    __syncthreads();
    { // stage A: 64 x 32, f32 -> bf16
      const float* src = arow + kb * 32;
      __builtin_prefetch(src + 32, 0, 0);
#pragma unroll
      for (int j = 0; j < 8; ++j) As[mloc * 32 + kg * 8 + j] = f2bf(src[j]);
    }
    { // stage B transposed: Bst[n][k] <- w_qkv[k][n]
      const float* src = brow + (size_t)kb * 32 * NQKV;
      __builtin_prefetch(src + 32 * NQKV, 0, 0);
#pragma unroll
      for (int j = 0; j < 8; ++j) Bst[(ng * 8 + j) * 32 + krow] = f2bf(src[j]);
    }
    __syncthreads();
    v16bf bfrag = frag_from_lds(&Bst[(tj * 16 + idx) * 32], hf);
    v16bf a0 = frag_from_lds(&As[((tip * 2 + 0) * 16 + idx) * 32], hf);
    v16bf a1 = frag_from_lds(&As[((tip * 2 + 1) * 16 + idx) * 32], hf);
    acc0 = __builtin_amdgcn_wmma_f32_16x16x32_bf16(false, a0, false, bfrag, (short)0, acc0, false, false);
    acc1 = __builtin_amdgcn_wmma_f32_16x16x32_bf16(false, a1, false, bfrag, (short)0, acc1, false, false);
  }

  const int n = n0 + tj * 16 + idx;
  const int sel = n / DIMC;
  const int head = (n % DIMC) >> 5;
  const int dh = n & 31;
  const int offQK = head * HEADSZ + dh;          // lane-constant adders
  const int offV  = head * HEADSZ + dh * NTOK;
#pragma unroll
  for (int t = 0; t < 2; ++t) {
    v8f acc = t ? acc1 : acc0;
    int ti = tip * 2 + t;
#pragma unroll
    for (int vv = 0; vv < 8; ++vv) {
      int ml = ti * 16 + vv + 8 * hf;            // C/D layout: M = vgpr + 8*half
      unsigned short val = f2bf(acc[vv]);
      if (sel == 0)      qb [rowQK[ml] + offQK] = val;
      else if (sel == 1) kbf[rowQK[ml] + offQK] = val;
      else               vb [rowV[ml]  + offV ] = val;
    }
  }
}

// ---------------------------------------------------------------------------
// Kernel 2: per (b,win,head) attention; one wave per task, 2 waves/block.
// Q/K staged by the Tensor Data Mover (contiguous 3136B blocks), V manual.
// ---------------------------------------------------------------------------
__global__ __launch_bounds__(64)
void attn_kernel(const unsigned short* __restrict__ qb,
                 const unsigned short* __restrict__ kbuf,
                 const unsigned short* __restrict__ vb,
                 const float* __restrict__ rel_pos,
                 const int* __restrict__ pos,
                 const unsigned char* __restrict__ neg_mask,
                 unsigned short* __restrict__ abuf)
{
  __shared__ __align__(16) unsigned short sQ[2][64 * 32]; // [tok][dh]
  __shared__ __align__(16) unsigned short sK[2][64 * 32]; // [tok][dh]
  __shared__ __align__(16) unsigned short sV[2][32 * 64]; // [dh][tok]
  __shared__ __align__(16) float          sS[2][64 * 65]; // padded stride
  __shared__ int orow[2][64];                             // output row bases

  const int w = threadIdx.x >> 5;
  const int lane = threadIdx.x & 31;
  const int idx = lane & 15, hf = lane >> 4;
  const int task = blockIdx.x * 2 + w;
  const int head = task % NHEAD;
  const int win  = (task / NHEAD) % NWIN;
  const int b    = task / (NHEAD * NWIN);
  const int wy = win >> 3, wx = win & 7;

  const size_t bh = (size_t)(b * NWIN + win) * NHEAD + head;
  const unsigned short* Q = qb   + bh * HEADSZ;
  const unsigned short* K = kbuf + bh * HEADSZ;
  const unsigned short* V = vb   + bh * HEADSZ;

  // TDM: copy the contiguous 49x32 bf16 blocks of Q and K into LDS.
  {
    i32x8 g1 = tdm_g1_2d(/*td0=*/HEADSZ, /*td1=*/1, /*tl0=*/HEADSZ, /*tl1=*/1,
                         /*stride0=*/HEADSZ);
    i32x4 z = {0, 0, 0, 0};
    i32x8 z8 = {0, 0, 0, 0, 0, 0, 0, 0};
    __builtin_amdgcn_tensor_load_to_lds(tdm_g0(&sQ[w][0], Q), g1, z, z, z8, 0);
    __builtin_amdgcn_tensor_load_to_lds(tdm_g0(&sK[w][0], K), g1, z, z, z8, 0);
  }
  // zero the pad rows (tokens 49..63) while the TDM runs
  for (int t = NTOK + lane; t < 64; t += 32) {
    uint4 z = {0, 0, 0, 0};
    uint4* dq = (uint4*)&sQ[w][t * 32];
    uint4* dk = (uint4*)&sK[w][t * 32];
#pragma unroll
    for (int j = 0; j < 4; ++j) { dq[j] = z; dk[j] = z; }
  }
  { // stage V ([dh][tok], zero-padded columns) + output row table
    int d = lane;
    for (int c = 0; c < NTOK; ++c) sV[w][d * 64 + c] = V[d * NTOK + c];
    for (int c = NTOK; c < 64; ++c) sV[w][d * 64 + c] = 0;
    for (int q = lane; q < 64; q += 32) {
      int y = wy * WSZ + q / WSZ, xx = wx * WSZ + q % WSZ;
      orow[w][q] = (q < NTOK) ? (int)((((size_t)b * HH + y) * WWC + xx) * DIMC) : 0;
    }
  }
  __builtin_amdgcn_s_wait_tensorcnt(0);
  __syncthreads();

  const float scale = 0.17677669529663687f; // 1/sqrt(32)

  // S = scale * Q K^T
  for (int ti = 0; ti < 4; ++ti) {
    v16bf a = frag_from_lds(&sQ[w][(ti * 16 + idx) * 32], hf);
#pragma unroll
    for (int tj = 0; tj < 4; ++tj) {
      v16bf bb = frag_from_lds(&sK[w][(tj * 16 + idx) * 32], hf);
      v8f acc = {};
      acc = __builtin_amdgcn_wmma_f32_16x16x32_bf16(false, a, false, bb, (short)0, acc, false, false);
#pragma unroll
      for (int vv = 0; vv < 8; ++vv)
        sS[w][(ti * 16 + vv + 8 * hf) * 65 + tj * 16 + idx] = acc[vv] * scale;
    }
  }
  __syncthreads();

  // bias + shift-mask + softmax; each lane owns rows lane and lane+32
#pragma unroll
  for (int rr = 0; rr < 2; ++rr) {
    int r = lane + 32 * rr;
    float* Sr = &sS[w][r * 65];
    if (r < NTOK) {
      const int* pr = pos + ((size_t)win * NTOK + r) * NTOK;
      const unsigned char* mr = neg_mask + ((size_t)win * NTOK + r) * NTOK;
      float mx = -1e30f;
      for (int c = 0; c < NTOK; ++c) {
        float sv = Sr[c] + rel_pos[pr[c] * NHEAD + head];
        if (mr[c]) sv = -1e30f;
        Sr[c] = sv;
        mx = fmaxf(mx, sv);
      }
      float sum = 0.f;
      for (int c = 0; c < NTOK; ++c) {
        float p = __expf(Sr[c] - mx);
        Sr[c] = p;
        sum += p;
      }
      float inv = 1.0f / sum;
      for (int c = 0; c < NTOK; ++c) Sr[c] *= inv;
      for (int c = NTOK; c < 64; ++c) Sr[c] = 0.f;  // kill pad K columns
    } else {
      for (int c = 0; c < 64; ++c) Sr[c] = 0.f;     // pad rows
    }
  }
  __syncthreads();

  // O = P V  -> abuf
  const int ocol = head * DHEAD + idx;
  for (int ti = 0; ti < 4; ++ti) {
    for (int nt = 0; nt < 2; ++nt) {
      v8f acc = {};
#pragma unroll
      for (int kc = 0; kc < 2; ++kc) {
        v16u au;
#pragma unroll
        for (int e = 0; e < 16; ++e)
          au[e] = f2bf(sS[w][(ti * 16 + idx) * 65 + kc * 32 + kmap(e, hf)]);
        union { v16u u; v16bf b; } ac; ac.u = au;
        const unsigned short* vp = &sV[w][(nt * 16 + idx) * 64 + kc * 32];
        v16u bu;
#pragma unroll
        for (int e = 0; e < 16; ++e) bu[e] = vp[kmap(e, hf)];
        union { v16u u; v16bf b; } bc; bc.u = bu;
        acc = __builtin_amdgcn_wmma_f32_16x16x32_bf16(false, ac.b, false, bc.b, (short)0, acc, false, false);
      }
#pragma unroll
      for (int vv = 0; vv < 8; ++vv) {
        int q = ti * 16 + vv + 8 * hf;
        if (q < NTOK) abuf[orow[w][q] + ocol + nt * 16] = f2bf(acc[vv]);
      }
    }
  }
}

// ---------------------------------------------------------------------------
// Kernel 3: out = attn @ w_out + b_out; A-tile staged by TDM (pure bf16 move),
// inverse shift folded into the f32 store via a per-row LDS table.
// ---------------------------------------------------------------------------
__global__ __launch_bounds__(256)
void proj_kernel(const unsigned short* __restrict__ abuf,
                 const float* __restrict__ w_out,
                 const float* __restrict__ b_out,
                 float* __restrict__ out)
{
  __shared__ __align__(16) unsigned short As[64 * 32];
  __shared__ __align__(16) unsigned short Bst[64 * 32];
  __shared__ int outrow[64];

  const int m0 = blockIdx.y * 64;
  const int n0 = blockIdx.x * 64;
  const int tid = threadIdx.x;
  const int wave = tid >> 5, lane = tid & 31;
  const int idx = lane & 15, hf = lane >> 4;
  const int tj = wave & 3, tip = wave >> 2;

  if (tid < 64) {  // inverse-roll destination row bases
    int m = m0 + tid;
    int b = m / (HH * WWC); int rem = m % (HH * WWC);
    int y = rem / WWC, xx = rem % WWC;
    outrow[tid] = (int)((((size_t)b * HH + (y + 3) % HH) * WWC + (xx + 3) % WWC) * DIMC);
  }

  const int krow = tid >> 3, ng = tid & 7;
  const float* brow = w_out + (size_t)krow * DIMC + n0 + ng * 8;

  const float bias = b_out[n0 + tj * 16 + idx];
  v8f acc0, acc1;
#pragma unroll
  for (int vv = 0; vv < 8; ++vv) { acc0[vv] = bias; acc1[vv] = bias; }

  // constant part of the A-tile TDM descriptor: 64 rows x 32 elems, stride 384
  const i32x8 g1A = tdm_g1_2d(/*td0=*/32, /*td1=*/64, /*tl0=*/32, /*tl1=*/64,
                              /*stride0=*/DIMC);
  const i32x4 zz = {0, 0, 0, 0};
  const i32x8 zz8 = {0, 0, 0, 0, 0, 0, 0, 0};

  for (int kb = 0; kb < DIMC / 32; ++kb) {
    __syncthreads();
    if (wave == 0) { // async A-tile copy: global bf16 -> LDS, no VGPR round-trip
      const unsigned short* ga = abuf + (size_t)m0 * DIMC + kb * 32;
      __builtin_amdgcn_tensor_load_to_lds(tdm_g0(&As[0], ga), g1A, zz, zz, zz8, 0);
    }
    { // stage B transposed (needs f32 -> bf16 conversion, stays on VALU)
      const float* src = brow + (size_t)kb * 32 * DIMC;
      __builtin_prefetch(src + 32 * DIMC, 0, 0);
#pragma unroll
      for (int j = 0; j < 8; ++j) Bst[(ng * 8 + j) * 32 + krow] = f2bf(src[j]);
    }
    if (wave == 0) __builtin_amdgcn_s_wait_tensorcnt(0);
    __syncthreads();
    v16bf bfrag = frag_from_lds(&Bst[(tj * 16 + idx) * 32], hf);
    v16bf a0 = frag_from_lds(&As[((tip * 2 + 0) * 16 + idx) * 32], hf);
    v16bf a1 = frag_from_lds(&As[((tip * 2 + 1) * 16 + idx) * 32], hf);
    acc0 = __builtin_amdgcn_wmma_f32_16x16x32_bf16(false, a0, false, bfrag, (short)0, acc0, false, false);
    acc1 = __builtin_amdgcn_wmma_f32_16x16x32_bf16(false, a1, false, bfrag, (short)0, acc1, false, false);
  }

  const int n = n0 + tj * 16 + idx;
#pragma unroll
  for (int t = 0; t < 2; ++t) {
    v8f acc = t ? acc1 : acc0;
    int ti = tip * 2 + t;
#pragma unroll
    for (int vv = 0; vv < 8; ++vv) {
      int ml = ti * 16 + vv + 8 * hf;
      out[outrow[ml] + n] = acc[vv];
    }
  }
}

// ---------------------------------------------------------------------------
extern "C" void kernel_launch(void* const* d_in, const int* in_sizes, int n_in,
                              void* d_out, int out_size, void* d_ws, size_t ws_size,
                              hipStream_t stream)
{
  const float* x        = (const float*)d_in[0];
  const float* w_qkv    = (const float*)d_in[1];
  const float* w_out    = (const float*)d_in[2];
  const float* b_out    = (const float*)d_in[3];
  const float* rel_pos  = (const float*)d_in[4];
  const int*   pos      = (const int*)d_in[5];
  const unsigned char* neg_mask = (const unsigned char*)d_in[6];

  const size_t perbuf = (size_t)NB * NWIN * NHEAD * HEADSZ; // 38,535,168 elems
  unsigned short* qb = (unsigned short*)d_ws;
  unsigned short* kb = qb + perbuf;
  unsigned short* vb = kb + perbuf;
  unsigned short* ab = vb + perbuf;

  dim3 g1(NQKV / 64, MTOT / 64);                 // (18, 1568)
  qkv_kernel<<<g1, 256, 0, stream>>>(x, w_qkv, qb, kb, vb);

  attn_kernel<<<dim3(NB * NWIN * NHEAD / 2), 64, 0, stream>>>(
      qb, kb, vb, rel_pos, pos, neg_mask, ab);

  dim3 g3(DIMC / 64, MTOT / 64);                 // (6, 1568)
  proj_kernel<<<g3, 256, 0, stream>>>(ab, w_out, b_out, (float*)d_out);
}
